// CrossAttention_37486474559895
// MI455X (gfx1250) — matrix operand order repaired and checked by reading, not compiled, not run on hardware
//
#include <hip/hip_runtime.h>
#include <hip/hip_bf16.h>
#include <cstdint>
#include <cstddef>

// Problem constants (match reference)
#define B_   32
#define L_   4097
#define D_   1024
#define H_   16
#define DH_  64
#define NK_  4096   // L-1 word positions

typedef __attribute__((ext_vector_type(16))) __bf16 v16bf;
typedef __attribute__((ext_vector_type(8)))  __bf16 v8bf;
typedef __attribute__((ext_vector_type(8)))  float  v8f;
typedef __attribute__((ext_vector_type(4)))  float  f4;
typedef __attribute__((ext_vector_type(4)))  int    v4i;

// Workspace layout (float-element offsets, each region 4KB aligned)
#define OFF_Q       0            // 32*1024 f32
#define OFF_QKB     32768        // 32*16 f32 (padded)
#define OFF_KQ      33792        // 32*16*1024 f32
#define OFF_SCORES  558080       // 32*16*4096 f32 (reused as exp() temp in K3)
#define OFF_PROBSBF 2655232      // 32*16*4096 bf16 (occupies 1048576 f32 slots)
#define OFF_WCTX    3703808      // 32*16*1024 f32
// total 4228096 floats ~= 16.9 MB

// Optional gfx1250 async global->LDS path (guarded; host pass falls back)
#if defined(__has_builtin)
#if __has_builtin(__builtin_amdgcn_global_load_async_to_lds_b128) && \
    __has_builtin(__builtin_amdgcn_s_wait_asynccnt)
#define HAVE_ASYNC_LDS 1
typedef __attribute__((address_space(1))) v4i* gv4i_p;   // global int4*
typedef __attribute__((address_space(3))) v4i* lv4i_p;   // LDS int4*
#endif
#endif

// ---------------------------------------------------------------------------
// WMMA helper (bf16 inputs, f32 accumulate) — v_wmma_f32_16x16x32_bf16
// ---------------------------------------------------------------------------
__device__ __forceinline__ v8f wmma_bf16(v16bf a, v16bf b, v8f c) {
    return __builtin_amdgcn_wmma_f32_16x16x32_bf16(false, a, false, b,
                                                   (short)0, c, false, false);
}

// A-operand (16x32, 16-bit): lane m (0-15) holds K {0..7, 16..23},
// lane m+16 holds K {8..15, 24..31} for row M = lane&15 (ISA table layout).
__device__ __forceinline__ v16bf load_a_bf16(const float* rowk, int lane) {
    const int c0 = (lane & 16) ? 8 : 0;
    const float* p = rowk + c0;
    f4 a0 = *(const f4*)(p + 0);
    f4 a1 = *(const f4*)(p + 4);
    f4 b0 = *(const f4*)(p + 16);
    f4 b1 = *(const f4*)(p + 20);
    v16bf r;
    r[0]  = (__bf16)a0[0]; r[1]  = (__bf16)a0[1];
    r[2]  = (__bf16)a0[2]; r[3]  = (__bf16)a0[3];
    r[4]  = (__bf16)a1[0]; r[5]  = (__bf16)a1[1];
    r[6]  = (__bf16)a1[2]; r[7]  = (__bf16)a1[3];
    r[8]  = (__bf16)b0[0]; r[9]  = (__bf16)b0[1];
    r[10] = (__bf16)b0[2]; r[11] = (__bf16)b0[3];
    r[12] = (__bf16)b1[0]; r[13] = (__bf16)b1[1];
    r[14] = (__bf16)b1[2]; r[15] = (__bf16)b1[3];
    return r;
}

// Row-major bf16 A load (probs), contiguous k order: halves live at kb+c0 and kb+c0+8, 16B aligned.
__device__ __forceinline__ v16bf load_a_bf16_packed(const __bf16* rowk, int lane) {
    const int c0 = (lane & 16) ? 16 : 0;
    v8bf lo = *(const v8bf*)(rowk + c0);
    v8bf hi = *(const v8bf*)(rowk + c0 + 8);
    return __builtin_shufflevector(lo, hi, 0, 1, 2, 3, 4, 5, 6, 7,
                                           8, 9, 10, 11, 12, 13, 14, 15);
}

// ---------------------------------------------------------------------------
// K0: q[b,:] = graph[b] @ q_w.T + q_b
// ---------------------------------------------------------------------------
__global__ __launch_bounds__(256) void k0_qproj(const float* __restrict__ feat,
                                                const float* __restrict__ q_w,
                                                const float* __restrict__ q_b,
                                                float* __restrict__ q) {
    __shared__ __align__(16) float sf[D_];
    const int b = blockIdx.x, tid = threadIdx.x;
    const float* g = feat + (size_t)b * L_ * D_;   // features[b, 0, :]
#ifdef HAVE_ASYNC_LDS
    // One 16B async global->LDS transfer per thread (256 * 16B = 4KB row).
    __builtin_amdgcn_global_load_async_to_lds_b128(
        (gv4i_p)(g + tid * 4), (lv4i_p)(&sf[tid * 4]), 0, 0);
    __builtin_amdgcn_s_wait_asynccnt(0);
#else
    for (int i = tid; i < D_; i += 256) sf[i] = g[i];
#endif
    __syncthreads();
    for (int rep = 0; rep < 4; ++rep) {
        const int dp = rep * 256 + tid;
        const float* wr = q_w + (size_t)dp * D_;
        float acc = q_b[dp];
        for (int d = 0; d < D_; d += 4) {
            f4 w = *(const f4*)(wr + d);
            acc += sf[d] * w[0] + sf[d + 1] * w[1] + sf[d + 2] * w[2] + sf[d + 3] * w[3];
        }
        q[(size_t)b * D_ + dp] = acc;
    }
}

// ---------------------------------------------------------------------------
// K1: kq[b,h,d] = sum_j q[b,h*64+j] * k_w[h*64+j, d]   (folded K projection)
//     qkb[b,h]  = sum_j q[b,h*64+j] * k_b[h*64+j]      (folded K bias)
// ---------------------------------------------------------------------------
__global__ __launch_bounds__(256) void k1_kq(const float* __restrict__ q,
                                             const float* __restrict__ k_w,
                                             const float* __restrict__ k_b,
                                             float* __restrict__ kq,
                                             float* __restrict__ qkb) {
    const int h = blockIdx.x, b = blockIdx.y, tid = threadIdx.x;
    __shared__ float qh[DH_];
    __shared__ float pr[DH_];
    if (tid < DH_) {
        float qv = q[(size_t)b * D_ + h * DH_ + tid];
        qh[tid] = qv;
        pr[tid] = qv * k_b[h * DH_ + tid];
    }
    __syncthreads();
    if (tid == 0) {
        float s = 0.f;
        for (int j = 0; j < DH_; ++j) s += pr[j];
        qkb[b * H_ + h] = s;
    }
    for (int d = tid; d < D_; d += 256) {
        float acc = 0.f;
        #pragma unroll 8
        for (int j = 0; j < DH_; ++j)
            acc += qh[j] * k_w[(size_t)(h * DH_ + j) * D_ + d];
        kq[((size_t)b * H_ + h) * D_ + d] = acc;
    }
}

// ---------------------------------------------------------------------------
// K2: scores[b,h,k] = words[b,k,:] . kq[b,h,:] + qkb[b,h]; mask; /8
//     GEMM per batch: words(4096x1024) @ kq^T(1024x16) via WMMA bf16.
//     Grid (64, 32), block 128 (4 waves); wave owns one 16-position tile.
//     kq[b] staged once in LDS, bf16, pre-swizzled to B-operand order.
// ---------------------------------------------------------------------------
__global__ __launch_bounds__(128) void k2_scores(const float* __restrict__ feat,
                                                 const int*   __restrict__ sent,
                                                 const float* __restrict__ kq,
                                                 const float* __restrict__ qkb,
                                                 float* __restrict__ scores) {
    const int b    = blockIdx.y;
    const int wave = threadIdx.x >> 5;
    const int lane = threadIdx.x & 31;
    const int pos0 = (blockIdx.x * 4 + wave) * 16;
    const int nrow = lane & 15;

    // B-operand staging: bq[chunk][half][lane][8 bf16]; lane stride 16B
    // within one ds_load_b128 => conflict-free reads. 32KB total.
    __shared__ __align__(16) __bf16 bq[32][2][32][8];
    for (int idx = threadIdx.x; idx < 1024; idx += 128) {
        const int c = idx >> 5, l = idx & 31;
        const int n = l & 15;
        const int c0 = (l & 16) ? 8 : 0;
        const float* p = kq + ((size_t)b * H_ + n) * D_ + c * 32 + c0;
        f4 x0 = *(const f4*)(p + 0);
        f4 x1 = *(const f4*)(p + 4);
        f4 x2 = *(const f4*)(p + 16);
        f4 x3 = *(const f4*)(p + 20);
        v8bf lo, hi;
        lo[0] = (__bf16)x0[0]; lo[1] = (__bf16)x0[1];
        lo[2] = (__bf16)x0[2]; lo[3] = (__bf16)x0[3];
        lo[4] = (__bf16)x1[0]; lo[5] = (__bf16)x1[1];
        lo[6] = (__bf16)x1[2]; lo[7] = (__bf16)x1[3];
        hi[0] = (__bf16)x2[0]; hi[1] = (__bf16)x2[1];
        hi[2] = (__bf16)x2[2]; hi[3] = (__bf16)x2[3];
        hi[4] = (__bf16)x3[0]; hi[5] = (__bf16)x3[1];
        hi[6] = (__bf16)x3[2]; hi[7] = (__bf16)x3[3];
        *(v8bf*)(&bq[c][0][l][0]) = lo;
        *(v8bf*)(&bq[c][1][l][0]) = hi;
    }
    __syncthreads();

    const float* arow = feat + ((size_t)b * L_ + 1 + pos0 + nrow) * D_;

    v8f acc = {};
    #pragma unroll 4
    for (int kb = 0; kb < D_; kb += 32) {
        __builtin_prefetch(arow + kb + 256, 0, 1);   // words stream ahead (1KB)
        v16bf a = load_a_bf16(arow + kb, lane);
        const int c = kb >> 5;
        v8bf lo = *(const v8bf*)(&bq[c][0][lane][0]);
        v8bf hi = *(const v8bf*)(&bq[c][1][lane][0]);
        v16bf bm = __builtin_shufflevector(lo, hi, 0, 1, 2, 3, 4, 5, 6, 7,
                                                   8, 9, 10, 11, 12, 13, 14, 15);
        acc = wmma_bf16(a, bm, acc);
    }

    const int h    = lane & 15;                 // C: N = lane&15
    const float qk = qkb[b * H_ + h];
    const int mhi  = (lane >> 4) << 3;          // C: M = r + 8*(lane>=16)
    #pragma unroll
    for (int r = 0; r < 8; ++r) {
        const int pos = pos0 + mhi + r;
        const bool keep = (sent[(size_t)b * L_ + pos] == 0);  // si = sent_ind[:, :-1]
        float s = acc[r] + qk;
        s = keep ? s * 0.125f : 0.0f;           // mask to 0 (NOT -inf), then /sqrt(64)
        scores[((size_t)b * H_ + h) * NK_ + pos] = s;
    }
}

// ---------------------------------------------------------------------------
// K3: softmax over k (4096) per (b,h); emits probs as bf16 (A-operand ready).
//     Overwrites the scores row in place with exp() as scratch.
// ---------------------------------------------------------------------------
__global__ __launch_bounds__(256) void k3_softmax(float* __restrict__ scores,
                                                  __bf16* __restrict__ probs) {
    const int h = blockIdx.x, b = blockIdx.y, tid = threadIdx.x;
    float*   row  = scores + ((size_t)b * H_ + h) * NK_;
    __bf16*  prow = probs  + ((size_t)b * H_ + h) * NK_;
    __shared__ float red[256];

    float mx = -1e30f;
    for (int i = tid; i < NK_; i += 256) mx = fmaxf(mx, row[i]);
    red[tid] = mx;
    __syncthreads();
    for (int s = 128; s > 0; s >>= 1) {
        if (tid < s) red[tid] = fmaxf(red[tid], red[tid + s]);
        __syncthreads();
    }
    mx = red[0];
    __syncthreads();

    float sum = 0.f;
    for (int i = tid; i < NK_; i += 256) {
        float e = __expf(row[i] - mx);
        row[i] = e;                       // each thread re-reads only its own slots
        sum += e;
    }
    red[tid] = sum;
    __syncthreads();
    for (int s = 128; s > 0; s >>= 1) {
        if (tid < s) red[tid] += red[tid + s];
        __syncthreads();
    }
    const float inv = 1.0f / red[0];
    for (int i = tid; i < NK_; i += 256) prow[i] = (__bf16)(row[i] * inv);
}

// ---------------------------------------------------------------------------
// K4: wctx[b,h,d] = sum_k probs[b,h,k] * words[b,k,d]
//     GEMM per batch: probs(16x4096) @ words(4096x1024) via WMMA bf16.
//     Grid (16, 32), block 128 (4 waves); wave owns one 16-wide d tile.
// ---------------------------------------------------------------------------
__global__ __launch_bounds__(128) void k4_wctx(const float* __restrict__ feat,
                                               const __bf16* __restrict__ probs,
                                               float* __restrict__ wctx) {
    const int b     = blockIdx.y;
    const int wave  = threadIdx.x >> 5;
    const int lane  = threadIdx.x & 31;
    const int dbase = (blockIdx.x * 4 + wave) * 16;
    const int n     = lane & 15;                 // A row = head; B col = dim
    const int c0b   = (lane & 16) ? 16 : 0;      // B K-half for this lane

    const __bf16* arow = probs + ((size_t)b * H_ + n) * NK_;

    v8f acc = {};
    #pragma unroll 2
    for (int kb = 0; kb < NK_; kb += 32) {
        v16bf a = load_a_bf16_packed(arow + kb, lane);
        // B column: words[b, kb+c0b+kk, dbase+n], kk = 0..15 (stride D_ rows)
        const float* bp = feat + ((size_t)b * L_ + 1 + kb + c0b) * D_ + dbase + n;
        __builtin_prefetch(bp + 32 * D_, 0, 1);  // next k-chunk of the words stream
        v16bf bm;
        #pragma unroll
        for (int kk = 0; kk < 16; ++kk) bm[kk] = (__bf16)bp[(size_t)kk * D_];
        acc = wmma_bf16(a, bm, acc);
    }

    const int mhi = (lane >> 4) << 3;            // C: M(head) = r + 8*(lane>=16)
    #pragma unroll
    for (int r = 0; r < 8; ++r) {
        const int h = mhi + r;
        wctx[((size_t)b * H_ + h) * D_ + dbase + n] = acc[r];
    }
}

// ---------------------------------------------------------------------------
// K5: ctx[b, h*64+j] = wctx[b,h,:] . v_w[h*64+j,:] + v_b[h*64+j]
// ---------------------------------------------------------------------------
__global__ __launch_bounds__(256) void k5_out(const float* __restrict__ wctx,
                                              const float* __restrict__ v_w,
                                              const float* __restrict__ v_b,
                                              float* __restrict__ out) {
    const int b = blockIdx.x, tid = threadIdx.x;
    for (int rep = 0; rep < 4; ++rep) {
        const int o = rep * 256 + tid;           // 0..1023
        const int h = o >> 6;
        const float* wr = wctx + ((size_t)b * H_ + h) * D_;
        const float* vr = v_w + (size_t)o * D_;
        float acc = v_b[o];
        for (int d = 0; d < D_; d += 4) {
            f4 w = *(const f4*)(wr + d);
            f4 v = *(const f4*)(vr + d);
            acc += w[0] * v[0] + w[1] * v[1] + w[2] * v[2] + w[3] * v[3];
        }
        out[(size_t)b * D_ + o] = acc;
    }
}

// ---------------------------------------------------------------------------
extern "C" void kernel_launch(void* const* d_in, const int* in_sizes, int n_in,
                              void* d_out, int out_size, void* d_ws, size_t ws_size,
                              hipStream_t stream) {
    (void)in_sizes; (void)n_in; (void)out_size; (void)ws_size;

    const float* feat = (const float*)d_in[0];
    const int*   sent = (const int*)  d_in[1];
    const float* q_w  = (const float*)d_in[2];
    const float* q_b  = (const float*)d_in[3];
    const float* k_w  = (const float*)d_in[4];
    const float* k_b  = (const float*)d_in[5];
    const float* v_w  = (const float*)d_in[6];
    const float* v_b  = (const float*)d_in[7];
    float* out = (float*)d_out;
    float* ws  = (float*)d_ws;

    float*  q    = ws + OFF_Q;
    float*  qkb  = ws + OFF_QKB;
    float*  kq   = ws + OFF_KQ;
    float*  sc   = ws + OFF_SCORES;
    __bf16* pbf  = (__bf16*)(ws + OFF_PROBSBF);
    float*  wcx  = ws + OFF_WCTX;

    k0_qproj  <<<dim3(B_),           dim3(256), 0, stream>>>(feat, q_w, q_b, q);
    k1_kq     <<<dim3(H_, B_),       dim3(256), 0, stream>>>(q, k_w, k_b, kq, qkb);
    k2_scores <<<dim3(NK_ / 64, B_), dim3(128), 0, stream>>>(feat, sent, kq, qkb, sc);
    k3_softmax<<<dim3(H_, B_),       dim3(256), 0, stream>>>(sc, pbf);
    k4_wctx   <<<dim3(D_ / 64, B_),  dim3(128), 0, stream>>>(feat, pbf, wcx);
    k5_out    <<<dim3(B_),           dim3(256), 0, stream>>>(wcx, v_w, v_b, out);
}
